// LogSig_RNN_50921132261615
// MI455X (gfx1250) — matrix operand order repaired and checked by reading, not compiled, not run on hardware
//
#include <hip/hip_runtime.h>
#include <hip/hip_bf16.h>
#include <math.h>

typedef __bf16 bf16;
typedef __attribute__((ext_vector_type(16))) __bf16 v16bf;
typedef __attribute__((ext_vector_type(8)))  __bf16 bf16x8;
typedef __attribute__((ext_vector_type(8)))  float  v8f;

#define BATCH   32
#define T_LEN   600
#define C_CH    64
#define NSEG    50
#define HID     512
#define G4      2048      // 4*HID
#define F_IN    2144      // C + C + C*(C-1)/2
#define LOGSIG_A 2016
#define M_ROWS  1600      // NSEG*BATCH

__device__ __forceinline__ float sigmoidf_(float x) { return 1.0f / (1.0f + __expf(-x)); }

// ---- WMMA fragment loaders (contiguous k order k = 16*h + s, not the ISA table order; 16-bit A 16x32 / B 32x16) ----
// A (MxK=16x32), row-major source, K contiguous:
//   lanes 0-15 : M=lane,    elems 0-15 = K[k0..k0+15]
//   lanes 16-31: M=lane-16, elems 0-15 = K[k0+16..k0+31]
__device__ __forceinline__ v16bf load_a_frag(const bf16* __restrict__ base, int ld, int m0, int k0) {
    int lane = threadIdx.x & 31;
    int row  = m0 + (lane & 15);
    int grp  = lane >> 4;
    const bf16* p = base + (size_t)row * ld + k0 + grp * 16;
    bf16x8 lo = *(const bf16x8*)p;
    bf16x8 hi = *(const bf16x8*)(p + 8);
    v16bf a;
#pragma unroll
    for (int i = 0; i < 8; ++i) { a[i] = lo[i]; a[i + 8] = hi[i]; }
    return a;
}
// B (KxN=32x16) taken from row-major W[n][k] (i.e. B^T), K contiguous per n:
//   lane = N (mod 16); lanes 0-15 hold K[k0..k0+15], lanes 16-31 hold K[k0+16..k0+31]
__device__ __forceinline__ v16bf load_b_frag(const bf16* __restrict__ base, int ld, int n0, int k0) {
    int lane = threadIdx.x & 31;
    int col  = n0 + (lane & 15);
    int grp  = lane >> 4;
    const bf16* p = base + (size_t)col * ld + k0 + grp * 16;
    return *(const v16bf*)p;
}

// ---- fp32 -> bf16 weight conversion ----
__global__ void cvt_bf16_kernel(const float* __restrict__ src, bf16* __restrict__ dst, int n) {
    for (int i = blockIdx.x * blockDim.x + threadIdx.x; i < n; i += gridDim.x * blockDim.x)
        dst[i] = (bf16)src[i];
}

// ---- zero h0 (bf16) and cell state (f32) ----
__global__ void init_state_kernel(float* __restrict__ cc, bf16* __restrict__ h0, int n) {
    int i = blockIdx.x * blockDim.x + threadIdx.x;
    if (i < n) { cc[i] = 0.0f; h0[i] = (bf16)0.0f; }
}

// ---- depth-2 segmented log-signature features -> bf16 feats (row = s*32 + b) ----
__global__ __launch_bounds__(256) void feats_kernel(const float* __restrict__ x,
                                                    bf16* __restrict__ feats) {
    __shared__ float xs[16 * C_CH];      // segment points (len <= 13)
    __shared__ float Qs[C_CH * C_CH];    // Q[i][j]
    int r = blockIdx.x;                  // 0..1599
    int b = r & 31, s = r >> 5;
    const double step = 599.0 / 50.0;    // np.linspace(1,600,51) step; rint == Python round
    int t0 = (int)rint(1.0 + (double)s * step) - 1;
    int t1 = (int)rint(1.0 + (double)(s + 1) * step) - 1;
    int len = t1 - t0 + 1;               // number of points in segment (inclusive)
    int tid = threadIdx.x;

    for (int idx = tid; idx < len * C_CH; idx += 256) {
        int tt = idx >> 6, cch = idx & 63;
        xs[idx] = x[((size_t)b * T_LEN + t0 + tt) * C_CH + cch];
    }
    __syncthreads();

    // Q[i][j] = sum_t x_t[i] * (x_{t+1}[j] - x_t[j])
    {
        int base = tid * 16;
#pragma unroll 4
        for (int e = base; e < base + 16; ++e) {
            int i = e >> 6, j = e & 63;
            float q = 0.0f;
            for (int tt = 0; tt < len - 1; ++tt)
                q += xs[tt * 64 + i] * (xs[(tt + 1) * 64 + j] - xs[tt * 64 + j]);
            Qs[e] = q;
        }
    }
    __syncthreads();

    // feats = [inc(64) | A_triu(2016) | xa(64)]
    for (int idx = tid; idx < F_IN; idx += 256) {
        float val;
        if (idx < C_CH) {
            val = xs[(len - 1) * 64 + idx] - xs[idx];
        } else if (idx < C_CH + LOGSIG_A) {
            int e = idx - C_CH;               // row-major triu(k=1) index
            int i = 0, rem = e;
            while (rem >= 63 - i) { rem -= 63 - i; ++i; }
            int j = i + 1 + rem;
            float xai = xs[i], xaj = xs[j];
            float inci = xs[(len - 1) * 64 + i] - xai;
            float incj = xs[(len - 1) * 64 + j] - xaj;
            // A[i][j] = 0.5*((Q - xa (x) inc) - transpose)
            val = 0.5f * (Qs[i * 64 + j] - Qs[j * 64 + i] - xai * incj + xaj * inci);
        } else {
            val = xs[idx - (C_CH + LOGSIG_A)]; // xa
        }
        feats[(size_t)r * F_IN + idx] = (bf16)val;
    }
}

// ---- gx = feats @ w_ih^T + b_ih + b_hh ; M=1600, N=2048, K=2144 ----
// 1 wave -> 16x64 output (4 N-tiles reuse A fragment). K loop is software-pipelined
// with a 2-stage register rotation (unroll pinned to 1 so the allocator keeps
// load results in-place across the backedge instead of emitting mov chains):
//   load B<-step k+1 ; wmma(A = step k) ; load A<-step k+2 ; wmma(B = step k+1)
// 67 K-steps = preload(1) + 33 pair-iterations + epilogue(1).
__global__ __launch_bounds__(256) void gemm_gx_kernel(const bf16* __restrict__ feats,
                                                      const bf16* __restrict__ w,
                                                      const float* __restrict__ b_ih,
                                                      const float* __restrict__ b_hh,
                                                      float* __restrict__ gx) {
    int wid = blockIdx.x * 8 + (threadIdx.x >> 5);  // 0..3199
    int mt = wid >> 5;                              // 0..99
    int ng = wid & 31;                              // 0..31
    int m0 = mt * 16, n0 = ng * 64;
    v8f c[4];
#pragma unroll
    for (int t = 0; t < 4; ++t)
#pragma unroll
        for (int v = 0; v < 8; ++v) c[t][v] = 0.0f;

    // preload K-step 0 into buffer A
    v16bf aA = load_a_frag(feats, F_IN, m0, 0);
    v16bf bA[4];
#pragma unroll
    for (int t = 0; t < 4; ++t) bA[t] = load_b_frag(w, F_IN, n0 + t * 16, 0);

#pragma unroll 1
    for (int it = 0; it < 33; ++it) {
        int k1 = 32 + it * 64;           // odd step
        int k2 = k1 + 32;                // even step (max 2112 = F_IN-32)
        // pull far-ahead weight lines toward the WGP (global_prefetch_b8)
        if (k1 + 128 < F_IN) {
            __builtin_prefetch(feats + (size_t)(m0 + (threadIdx.x & 15)) * F_IN + k1 + 128, 0, 1);
            __builtin_prefetch(w + (size_t)(n0 + (threadIdx.x & 31)) * F_IN + k1 + 128, 0, 1);
        }
        // load buffer B <- step k1, consume buffer A (step k1-32)
        v16bf aB = load_a_frag(feats, F_IN, m0, k1);
        v16bf bB[4];
#pragma unroll
        for (int t = 0; t < 4; ++t) bB[t] = load_b_frag(w, F_IN, n0 + t * 16, k1);
#pragma unroll
        for (int t = 0; t < 4; ++t)
            c[t] = __builtin_amdgcn_wmma_f32_16x16x32_bf16(false, aA, false, bA[t],
                                                           (short)0, c[t], false, false);
        // load buffer A <- step k2, consume buffer B (step k1)
        aA = load_a_frag(feats, F_IN, m0, k2);
#pragma unroll
        for (int t = 0; t < 4; ++t) bA[t] = load_b_frag(w, F_IN, n0 + t * 16, k2);
#pragma unroll
        for (int t = 0; t < 4; ++t)
            c[t] = __builtin_amdgcn_wmma_f32_16x16x32_bf16(false, aB, false, bB[t],
                                                           (short)0, c[t], false, false);
    }
    // epilogue: buffer A holds step 66 (k0 = 2112)
#pragma unroll
    for (int t = 0; t < 4; ++t)
        c[t] = __builtin_amdgcn_wmma_f32_16x16x32_bf16(false, aA, false, bA[t],
                                                       (short)0, c[t], false, false);

    int lane = threadIdx.x & 31, grp = lane >> 4, cl = lane & 15;
#pragma unroll
    for (int t = 0; t < 4; ++t) {
        int col = n0 + t * 16 + cl;
        float bias = b_ih[col] + b_hh[col];
#pragma unroll
        for (int v = 0; v < 8; ++v) {
            int row = m0 + v + 8 * grp;
            gx[(size_t)row * G4 + col] = c[t][v] + bias;
        }
    }
}

// ---- one LSTM step: g = gx_s + h_in @ w_hh^T ; gates; cc,h update ----
// wave owns (m-tile in [0,2), j-tile in [0,32)); 4 gate accumulators share A fragment.
// 16 K-steps = preload(1) + 7 pair-iterations (fully unrolled) + 1 pipelined + epilogue.
__global__ __launch_bounds__(256) void lstm_step_kernel(const float* __restrict__ gx_s,
                                                        const bf16* __restrict__ h_in,
                                                        const bf16* __restrict__ w_hh,
                                                        float* __restrict__ cc,
                                                        bf16* __restrict__ h_out,
                                                        float* __restrict__ out, int s) {
    int wid = blockIdx.x * 8 + (threadIdx.x >> 5);  // 0..63
    int mt = wid >> 5, jt = wid & 31;
    int m0 = mt * 16, j0 = jt * 16;
    int lane = threadIdx.x & 31, grp = lane >> 4, cl = lane & 15;

    v8f c[4];
#pragma unroll
    for (int g = 0; g < 4; ++g)
#pragma unroll
        for (int v = 0; v < 8; ++v)
            c[g][v] = gx_s[(size_t)(m0 + v + 8 * grp) * G4 + g * HID + j0 + cl];

    // preload K-step 0 into buffer A
    v16bf aA = load_a_frag(h_in, HID, m0, 0);
    v16bf bA[4];
#pragma unroll
    for (int g = 0; g < 4; ++g) bA[g] = load_b_frag(w_hh, HID, g * HID + j0, 0);

#pragma unroll
    for (int it = 0; it < 7; ++it) {
        int k1 = 32 + it * 64;
        int k2 = k1 + 32;
        v16bf aB = load_a_frag(h_in, HID, m0, k1);
        v16bf bB[4];
#pragma unroll
        for (int g = 0; g < 4; ++g) bB[g] = load_b_frag(w_hh, HID, g * HID + j0, k1);
#pragma unroll
        for (int g = 0; g < 4; ++g)
            c[g] = __builtin_amdgcn_wmma_f32_16x16x32_bf16(false, aA, false, bA[g],
                                                           (short)0, c[g], false, false);
        aA = load_a_frag(h_in, HID, m0, k2);
#pragma unroll
        for (int g = 0; g < 4; ++g) bA[g] = load_b_frag(w_hh, HID, g * HID + j0, k2);
#pragma unroll
        for (int g = 0; g < 4; ++g)
            c[g] = __builtin_amdgcn_wmma_f32_16x16x32_bf16(false, aB, false, bB[g],
                                                           (short)0, c[g], false, false);
    }
    {   // step 15 load, step 14 compute (buffer A holds step 14 after the pair loop)
        v16bf aB = load_a_frag(h_in, HID, m0, 480);
        v16bf bB[4];
#pragma unroll
        for (int g = 0; g < 4; ++g) bB[g] = load_b_frag(w_hh, HID, g * HID + j0, 480);
#pragma unroll
        for (int g = 0; g < 4; ++g)
            c[g] = __builtin_amdgcn_wmma_f32_16x16x32_bf16(false, aA, false, bA[g],
                                                           (short)0, c[g], false, false);
#pragma unroll
        for (int g = 0; g < 4; ++g)
            c[g] = __builtin_amdgcn_wmma_f32_16x16x32_bf16(false, aB, false, bB[g],
                                                           (short)0, c[g], false, false);
    }

#pragma unroll
    for (int v = 0; v < 8; ++v) {
        int m = m0 + v + 8 * grp;   // batch index
        int j = j0 + cl;            // hidden index
        float iv = sigmoidf_(c[0][v]);
        float fv = sigmoidf_(c[1][v]);
        float gv = tanhf(c[2][v]);
        float ov = sigmoidf_(c[3][v]);
        float cn = fv * cc[m * HID + j] + iv * gv;
        cc[m * HID + j] = cn;
        float h = ov * tanhf(cn);
        h_out[m * HID + j] = (bf16)h;
        out[((size_t)m * NSEG + s) * HID + j] = h;   // out[b, s, :]
    }
}

extern "C" void kernel_launch(void* const* d_in, const int* in_sizes, int n_in,
                              void* d_out, int out_size, void* d_ws, size_t ws_size,
                              hipStream_t stream) {
    const float* x    = (const float*)d_in[0];
    const float* w_ih = (const float*)d_in[1];
    const float* w_hh = (const float*)d_in[2];
    const float* b_ih = (const float*)d_in[3];
    const float* b_hh = (const float*)d_in[4];
    float* out = (float*)d_out;

    char* ws = (char*)d_ws;
    size_t off = 0;
    auto alloc = [&](size_t bytes) -> void* {
        void* p = ws + off;
        off += (bytes + 255) & ~(size_t)255;
        return p;
    };
    bf16*  feats   = (bf16*) alloc((size_t)M_ROWS * F_IN * 2);   // 6.9 MB
    bf16*  w_ih_bf = (bf16*) alloc((size_t)G4 * F_IN * 2);       // 8.8 MB
    bf16*  w_hh_bf = (bf16*) alloc((size_t)G4 * HID * 2);        // 2.1 MB
    float* gx      = (float*)alloc((size_t)M_ROWS * G4 * 4);     // 13.1 MB
    bf16*  hbuf0   = (bf16*) alloc((size_t)BATCH * HID * 2);
    bf16*  hbuf1   = (bf16*) alloc((size_t)BATCH * HID * 2);
    float* cc      = (float*)alloc((size_t)BATCH * HID * 4);

    cvt_bf16_kernel<<<2048, 256, 0, stream>>>(w_ih, w_ih_bf, G4 * F_IN);
    cvt_bf16_kernel<<<1024, 256, 0, stream>>>(w_hh, w_hh_bf, G4 * HID);
    feats_kernel<<<M_ROWS, 256, 0, stream>>>(x, feats);
    gemm_gx_kernel<<<400, 256, 0, stream>>>(feats, w_ih_bf, b_ih, b_hh, gx);
    init_state_kernel<<<64, 256, 0, stream>>>(cc, hbuf0, BATCH * HID);

    bf16* hin = hbuf0;
    bf16* hout = hbuf1;
    for (int s = 0; s < NSEG; ++s) {
        lstm_step_kernel<<<8, 256, 0, stream>>>(gx + (size_t)s * BATCH * G4,
                                                hin, w_hh_bf, cc, hout, out, s);
        bf16* t = hin; hin = hout; hout = t;
    }
}
